// GIN_35459249995961
// MI455X (gfx1250) — matrix operand order repaired and checked by reading, not compiled, not run on hardware
//
#include <hip/hip_runtime.h>

#define BN_EPS 1e-5f

typedef __bf16 bf16;
typedef __attribute__((ext_vector_type(16))) __bf16 v16bf;
typedef __attribute__((ext_vector_type(2)))  __bf16 bf16x2;
typedef __attribute__((ext_vector_type(8)))  float  v8f;

// ---------------- helpers ----------------

static __device__ __forceinline__ bf16 f2bf(float f) {
  unsigned u = __builtin_bit_cast(unsigned, f);
  u += 0x7FFFu + ((u >> 16) & 1u);          // round-to-nearest-even
  unsigned short s = (unsigned short)(u >> 16);
  return __builtin_bit_cast(bf16, s);
}

union AFrag { v16bf v; bf16x2 p[8]; };

// load one 16x32 bf16 A fragment (per ISA 16-bit A 16x32 VGPR layout)
static __device__ __forceinline__ v16bf load_a_frag(const bf16* Arow, int kb, int koff) {
  AFrag af;
  const bf16x2* pa0 = (const bf16x2*)(Arow + kb + koff);        // K = kb+koff .. +7
  const bf16x2* pa1 = (const bf16x2*)(Arow + kb + 16 + koff);   // K = kb+16+koff .. +7
#pragma unroll
  for (int t = 0; t < 4; t++) af.p[t]     = pa0[t];
#pragma unroll
  for (int t = 0; t < 4; t++) af.p[4 + t] = pa1[t];
  return af.v;
}

// ---------------- kernels ----------------

// f32 -> bf16 conversion
__global__ void k_f2bf(const float* __restrict__ src, bf16* __restrict__ dst, long n) {
  long i = (long)blockIdx.x * blockDim.x + threadIdx.x;
  if (i < n) dst[i] = f2bf(src[i]);
}

// scatter-add aggregation: agg[dst] += h[src]  (agg pre-initialized with h -> z = h + sum_j h_j)
// one thread per (edge, 4 features); lanes of a wave cover consecutive features of one edge
__global__ void k_edge_agg(const int* __restrict__ ei, const float* __restrict__ h,
                           float* __restrict__ agg, int E, int d) {
  long idx = (long)blockIdx.x * blockDim.x + threadIdx.x;
  int per = d >> 2;
  if (idx >= (long)E * per) return;
  int e = (int)(idx / per);
  int c = (int)(idx % per) * 4;
  int s = ei[e];
  int t = ei[E + e];
  const float4 v = *(const float4*)(h + (long)s * d + c);
  float* p = agg + (long)t * d + c;
  atomicAdd(p + 0, v.x);
  atomicAdd(p + 1, v.y);
  atomicAdd(p + 2, v.z);
  atomicAdd(p + 3, v.w);
}

// C[M,N] = A[M,K](bf16,row-major) x B[K,N](bf16,row-major) + bias; optional ReLU;
// optional f32 and/or bf16 outputs. Each wave computes a 32x32 tile (2x2 WMMA tiles,
// 4 accumulators) -> 4 v_wmma per 4 fragment loads. Block = 8 waves = 128x64 tile.
__global__ __launch_bounds__(256) void k_gemm_bf16(
    const bf16* __restrict__ A, const bf16* __restrict__ B,
    const float* __restrict__ bias,
    float* __restrict__ Cf, bf16* __restrict__ Cb,
    int M, int K, int Ncols, int relu)
{
  const int lane  = threadIdx.x & 31;
  const int w     = threadIdx.x >> 5;
  const int tileM = blockIdx.y * 128 + (w & 3) * 32;
  const int tileN = blockIdx.x * 64  + (w >> 2) * 32;
  if (tileM >= M) return;                       // wave-uniform; EXEC all-1s for WMMA
  const bool m1ok = (tileM + 16) < M;           // second M sub-tile valid? (M % 16 == 0)

  v8f acc00 = {0.f,0.f,0.f,0.f,0.f,0.f,0.f,0.f};
  v8f acc01 = acc00, acc10 = acc00, acc11 = acc00;

  // A fragments: lane holds row M = base + (lane&15); K packing per ISA 16-bit A 16x32 layout
  const int row0 = tileM + (lane & 15);
  const int row1 = m1ok ? (row0 + 16) : row0;   // safe duplicate loads on boundary wave
  const int koff = (lane < 16) ? 0 : 8;
  const bf16* Arow0 = A + (long)row0 * K;
  const bf16* Arow1 = A + (long)row1 * K;

  for (int kb = 0; kb < K; kb += 32) {
    v16bf a0 = load_a_frag(Arow0, kb, koff);
    v16bf a1 = load_a_frag(Arow1, kb, koff);

    // B fragments: lane holds N = base+(lane&15); K packing matches A (koff, koff+16)
    const bf16* Bcol = B + tileN + (lane & 15);
    v16bf b0, b1;
#pragma unroll
    for (int s = 0; s < 16; s++) {
      const long kr = (long)(kb + koff + (s < 8 ? s : s + 8)) * Ncols;
      b0[s] = Bcol[kr];
      b1[s] = Bcol[kr + 16];
    }

    if (kb + 32 < K) {
      __builtin_prefetch(Arow0 + kb + 32, 0, 1);   // global_prefetch_b8
      __builtin_prefetch(Arow1 + kb + 32, 0, 1);
    }

    acc00 = __builtin_amdgcn_wmma_f32_16x16x32_bf16(false, a0, false, b0, (short)0, acc00, false, false);
    acc01 = __builtin_amdgcn_wmma_f32_16x16x32_bf16(false, a0, false, b1, (short)0, acc01, false, false);
    acc10 = __builtin_amdgcn_wmma_f32_16x16x32_bf16(false, a1, false, b0, (short)0, acc10, false, false);
    acc11 = __builtin_amdgcn_wmma_f32_16x16x32_bf16(false, a1, false, b1, (short)0, acc11, false, false);
  }

  // D layout: lanes 0-15 -> N=lane, M=base+r ; lanes 16-31 -> N=lane-16, M=base+8+r
  const int n0 = tileN + (lane & 15);
  const int n1 = n0 + 16;
  const int r0 = tileM + ((lane < 16) ? 0 : 8);
  const float bv0 = bias ? bias[n0] : 0.0f;
  const float bv1 = bias ? bias[n1] : 0.0f;

#pragma unroll
  for (int r = 0; r < 8; r++) {
    const long rowOffA = (long)(r0 + r) * Ncols;
    float v00 = acc00[r] + bv0;
    float v01 = acc01[r] + bv1;
    if (relu) { v00 = fmaxf(v00, 0.0f); v01 = fmaxf(v01, 0.0f); }
    if (Cf) { Cf[rowOffA + n0] = v00; Cf[rowOffA + n1] = v01; }
    if (Cb) { Cb[rowOffA + n0] = f2bf(v00); Cb[rowOffA + n1] = f2bf(v01); }
    if (m1ok) {
      const long rowOffB = rowOffA + (long)16 * Ncols;
      float v10 = acc10[r] + bv0;
      float v11 = acc11[r] + bv1;
      if (relu) { v10 = fmaxf(v10, 0.0f); v11 = fmaxf(v11, 0.0f); }
      if (Cf) { Cf[rowOffB + n0] = v10; Cf[rowOffB + n1] = v11; }
      if (Cb) { Cb[rowOffB + n0] = f2bf(v10); Cb[rowOffB + n1] = f2bf(v11); }
    }
  }
}

// per-column sum / sum-of-squares (blockDim.x == d)
__global__ void k_colstats(const float* __restrict__ y, float* __restrict__ ssum,
                           float* __restrict__ ssq, int Nrows, int d, int rowsPerBlock) {
  int f  = threadIdx.x;
  int r0 = blockIdx.x * rowsPerBlock;
  int r1 = r0 + rowsPerBlock;
  if (r1 > Nrows) r1 = Nrows;
  float s = 0.f, ss = 0.f;
  for (int r = r0; r < r1; r++) {
    float v = y[(long)r * d + f];
    s += v; ss += v * v;
  }
  atomicAdd(&ssum[f], s);
  atomicAdd(&ssq[f], ss);
}

// xout = relu(BN(y)) + residual   (resid==nullptr -> identity residual: += old xout)
__global__ void k_bn_apply(const float* __restrict__ y, const float* __restrict__ ssum,
                           const float* __restrict__ ssq, const float* __restrict__ gamma,
                           const float* __restrict__ beta, const float* __restrict__ resid,
                           float* __restrict__ xout, int Nrows, int d) {
  long i = (long)blockIdx.x * blockDim.x + threadIdx.x;
  if (i >= (long)Nrows * d) return;
  int f = (int)(i % d);
  float inv = 1.0f / (float)Nrows;
  float m   = ssum[f] * inv;
  float var = ssq[f] * inv - m * m;
  float v = (y[i] - m) * rsqrtf(var + BN_EPS) * gamma[f] + beta[f];
  v = fmaxf(v, 0.0f);
  xout[i] = v + (resid ? resid[i] : xout[i]);
}

// log-softmax over C=64, one wave (32 lanes) per row, 2 values per lane
__global__ void k_logsoftmax64(const float* __restrict__ y, float* __restrict__ out, int Nrows) {
  int row  = blockIdx.x * (blockDim.x >> 5) + (threadIdx.x >> 5);
  int lane = threadIdx.x & 31;
  if (row >= Nrows) return;
  const float* p = y + (long)row * 64;
  float a = p[lane], b = p[lane + 32];
  float mx = fmaxf(a, b);
#pragma unroll
  for (int o = 16; o > 0; o >>= 1) mx = fmaxf(mx, __shfl_xor(mx, o, 32));
  float s = __expf(a - mx) + __expf(b - mx);
#pragma unroll
  for (int o = 16; o > 0; o >>= 1) s += __shfl_xor(s, o, 32);
  float ls = __logf(s);
  out[(long)row * 64 + lane]      = (a - mx) - ls;
  out[(long)row * 64 + lane + 32] = (b - mx) - ls;
}

// ---------------- host ----------------

static void gemm(hipStream_t s, const bf16* A, const bf16* B, const float* bias,
                 float* Cf, bf16* Cb, int M, int K, int Ncols, int relu) {
  dim3 g(Ncols / 64, (M + 127) / 128, 1);
  k_gemm_bf16<<<g, 256, 0, s>>>(A, B, bias, Cf, Cb, M, K, Ncols, relu);
}

extern "C" void kernel_launch(void* const* d_in, const int* in_sizes, int n_in,
                              void* d_out, int out_size, void* d_ws, size_t ws_size,
                              hipStream_t stream) {
  const int DIN = 128, H = 256, C = 64;
  const int N = in_sizes[0] / DIN;
  const int E = in_sizes[1] / 2;

  const float* x    = (const float*)d_in[0];
  const int*   ei   = (const int*)d_in[1];
  const float* W1_0 = (const float*)d_in[2];  const float* b1_0 = (const float*)d_in[3];
  const float* W2_0 = (const float*)d_in[4];  const float* b2_0 = (const float*)d_in[5];
  const float* g0   = (const float*)d_in[6];  const float* be0  = (const float*)d_in[7];
  const float* Wd_0 = (const float*)d_in[8];
  const float* W1_1 = (const float*)d_in[9];  const float* b1_1 = (const float*)d_in[10];
  const float* W2_1 = (const float*)d_in[11]; const float* b2_1 = (const float*)d_in[12];
  const float* g1   = (const float*)d_in[13]; const float* be1  = (const float*)d_in[14];
  const float* W1_2 = (const float*)d_in[15]; const float* b1_2 = (const float*)d_in[16];
  const float* W2_2 = (const float*)d_in[17]; const float* b2_2 = (const float*)d_in[18];

  // workspace carve-up (all chunks 256B-aligned)
  char* w = (char*)d_ws;
  float* z     = (float*)w; w += (size_t)N * H * 4;   // agg / z buffer
  float* y     = (float*)w; w += (size_t)N * H * 4;   // GIN output per layer
  float* xcur  = (float*)w; w += (size_t)N * H * 4;   // node features between layers
  float* resid = (float*)w; w += (size_t)N * H * 4;   // x @ Wd_0 (layer 0 only)
  bf16* ab     = (bf16*)w;  w += (size_t)N * H * 2;   // bf16 staging (GEMM A)
  bf16* hb     = (bf16*)w;  w += (size_t)N * H * 2;   // bf16 hidden activations
  bf16* W1_0b  = (bf16*)w;  w += (size_t)DIN * H * 2;
  bf16* Wd_0b  = (bf16*)w;  w += (size_t)DIN * H * 2;
  bf16* W2_0b  = (bf16*)w;  w += (size_t)H * H * 2;
  bf16* W1_1b  = (bf16*)w;  w += (size_t)H * H * 2;
  bf16* W2_1b  = (bf16*)w;  w += (size_t)H * H * 2;
  bf16* W1_2b  = (bf16*)w;  w += (size_t)H * H * 2;
  bf16* W2_2b  = (bf16*)w;  w += (size_t)H * C * 2;
  float* ssum  = (float*)w; w += H * 4;
  float* ssq   = (float*)w; w += H * 4;

  auto cvt = [&](const float* s, bf16* dptr, long n) {
    k_f2bf<<<dim3((unsigned)((n + 255) / 256)), 256, 0, stream>>>(s, dptr, n);
  };

  // quantize weights once per call
  cvt(W1_0, W1_0b, (long)DIN * H);  cvt(Wd_0, Wd_0b, (long)DIN * H);
  cvt(W2_0, W2_0b, (long)H * H);    cvt(W1_1, W1_1b, (long)H * H);
  cvt(W2_1, W2_1b, (long)H * H);    cvt(W1_2, W1_2b, (long)H * H);
  cvt(W2_2, W2_2b, (long)H * C);

  // ================= Layer 0 (d=128 in) =================
  hipMemcpyAsync(z, x, (size_t)N * DIN * 4, hipMemcpyDeviceToDevice, stream);
  { long nt = (long)E * (DIN / 4);
    k_edge_agg<<<dim3((unsigned)((nt + 255) / 256)), 256, 0, stream>>>(ei, x, z, E, DIN); }
  cvt(z, ab, (long)N * DIN);
  gemm(stream, ab, W1_0b, b1_0, nullptr, hb, N, DIN, H, /*relu=*/1);   // hidden (bf16)
  cvt(x, ab, (long)N * DIN);                                           // reuse ab = bf16(x)
  gemm(stream, ab, Wd_0b, nullptr, resid, nullptr, N, DIN, H, 0);      // residual
  gemm(stream, hb, W2_0b, b2_0, y, nullptr, N, H, H, 0);               // GIN out (f32)
  hipMemsetAsync(ssum, 0, H * 4, stream);
  hipMemsetAsync(ssq,  0, H * 4, stream);
  k_colstats<<<dim3((N + 127) / 128), H, 0, stream>>>(y, ssum, ssq, N, H, 128);
  k_bn_apply<<<dim3((unsigned)(((long)N * H + 255) / 256)), 256, 0, stream>>>(
      y, ssum, ssq, g0, be0, resid, xcur, N, H);

  // ================= Layer 1 (d=256) =================
  hipMemcpyAsync(z, xcur, (size_t)N * H * 4, hipMemcpyDeviceToDevice, stream);
  { long nt = (long)E * (H / 4);
    k_edge_agg<<<dim3((unsigned)((nt + 255) / 256)), 256, 0, stream>>>(ei, xcur, z, E, H); }
  cvt(z, ab, (long)N * H);
  gemm(stream, ab, W1_1b, b1_1, nullptr, hb, N, H, H, 1);
  gemm(stream, hb, W2_1b, b2_1, y, nullptr, N, H, H, 0);
  hipMemsetAsync(ssum, 0, H * 4, stream);
  hipMemsetAsync(ssq,  0, H * 4, stream);
  k_colstats<<<dim3((N + 127) / 128), H, 0, stream>>>(y, ssum, ssq, N, H, 128);
  k_bn_apply<<<dim3((unsigned)(((long)N * H + 255) / 256)), 256, 0, stream>>>(
      y, ssum, ssq, g1, be1, nullptr, xcur, N, H);                    // identity residual

  // ================= Layer 2 (final) =================
  hipMemcpyAsync(z, xcur, (size_t)N * H * 4, hipMemcpyDeviceToDevice, stream);
  { long nt = (long)E * (H / 4);
    k_edge_agg<<<dim3((unsigned)((nt + 255) / 256)), 256, 0, stream>>>(ei, xcur, z, E, H); }
  cvt(z, ab, (long)N * H);
  gemm(stream, ab, W1_2b, b1_2, nullptr, hb, N, H, H, 1);
  gemm(stream, hb, W2_2b, b2_2, y, nullptr, N, H, C, 0);              // [N, 64] logits
  k_logsoftmax64<<<dim3((N + 7) / 8), 256, 0, stream>>>(y, (float*)d_out, N);
}
